// GAT_833223655580
// MI455X (gfx1250) — matrix operand order repaired and checked by reading, not compiled, not run on hardware
//
#include <hip/hip_runtime.h>
#include <math.h>

typedef __attribute__((ext_vector_type(16))) _Float16 v16h;
typedef __attribute__((ext_vector_type(8)))  float    v8f;

#define NNODES 50000
#define NEDGES 800000
#define ETOT   (NNODES + NEDGES)
#define NEG_SLOPE 0.2f

// ---- order-preserving float <-> u32 keys for atomicMax-based segment max ----
__device__ __forceinline__ unsigned enc_f(float f) {
    unsigned u = __float_as_uint(f);
    return (u & 0x80000000u) ? ~u : (u | 0x80000000u);
}
__device__ __forceinline__ float dec_f(unsigned k) {
    if (k == 0u) return 0.0f;  // "no edge" -> matches reference's where(isfinite(m), m, 0)
    unsigned u = (k & 0x80000000u) ? (k & 0x7fffffffu) : ~k;
    return __uint_as_float(u);
}

// =====================================================================
// GEMM: H[N, NC] = X[N, 128] @ W[128, NC]   (f16 WMMA, f32 accumulate)
// One wave computes a 16-row stripe (NC/16 column tiles). W^T staged in
// LDS as f16 so each B fragment is one contiguous 32B read.
// =====================================================================
template <int NC>
__global__ __launch_bounds__(256) void gat_gemm(const float* __restrict__ X,
                                                const float* __restrict__ W,
                                                float* __restrict__ H,
                                                int nTiles) {
    __shared__ alignas(32) _Float16 ldsW[NC * 128];
    const int tid = threadIdx.x;
    // stage W transposed: ldsW[n*128 + k] = (f16) W[k*NC + n]
    for (int i = tid; i < NC * 128; i += 256) {
        int n = i >> 7, k = i & 127;
        ldsW[i] = (_Float16)W[k * NC + n];
    }
    __syncthreads();

    const int wave = tid >> 5, lane = tid & 31;
    const int tile = blockIdx.x * 8 + wave;         // wave-uniform guard -> EXEC all-1s
    if (tile >= nTiles) return;
    const int laneN = lane & 15, half = lane >> 4;
    const int row = tile * 16 + laneN;              // A-matrix M index
    constexpr int NT = NC / 16;

    v8f acc[NT];
#pragma unroll
    for (int t = 0; t < NT; t++) acc[t] = (v8f){0.f, 0.f, 0.f, 0.f, 0.f, 0.f, 0.f, 0.f};

#pragma unroll
    for (int kt = 0; kt < 4; kt++) {                // K = 128 = 4 x 32
        // A 16x32 f16 layout: lane holds K = kbase+half*16+(0..15)
        v16h a;
        const float* xr = X + (size_t)row * 128 + kt * 32 + half * 16;
#pragma unroll
        for (int i = 0; i < 8; i++) a[i] = (_Float16)xr[i];
#pragma unroll
        for (int i = 0; i < 8; i++) a[8 + i] = (_Float16)xr[8 + i];
#pragma unroll
        for (int nt = 0; nt < NT; nt++) {
            // B 32x16: lane holds col=laneN, K = kbase + half*16 + (0..15) -> contiguous in W^T
            const v16h* bp =
                (const v16h*)&ldsW[(nt * 16 + laneN) * 128 + kt * 32 + half * 16];
            acc[nt] = __builtin_amdgcn_wmma_f32_16x16x32_f16(
                false, a, false, *bp, (short)0, acc[nt], false, false);
        }
    }

    // C layout: VGPR r holds M = r (lanes 0-15) / 8+r (lanes 16-31), N = lane%16
    const int rbase = tile * 16 + half * 8;
#pragma unroll
    for (int nt = 0; nt < NT; nt++) {
        const int col = nt * 16 + laneN;
#pragma unroll
        for (int r = 0; r < 8; r++)
            H[(size_t)(rbase + r) * NC + col] = acc[nt][r];
    }
}

// ====================== attention logits per node ======================
template <int H_, int C_>
__global__ void gat_att(const float* __restrict__ Hf,
                        const float* __restrict__ att_s,
                        const float* __restrict__ att_d,
                        float* __restrict__ as_, float* __restrict__ ad_) {
    int t = blockIdx.x * blockDim.x + threadIdx.x;
    if (t >= NNODES * H_) return;
    int n = t / H_, hd = t % H_;
    const float* hrow = Hf + (size_t)n * (H_ * C_) + hd * C_;
    float ss = 0.f, sd = 0.f;
#pragma unroll 8
    for (int c = 0; c < C_; c++) {
        float v = hrow[c];
        ss += v * att_s[hd * C_ + c];
        sd += v * att_d[hd * C_ + c];
    }
    as_[t] = ss;
    ad_[t] = sd;
}

// ============ init: out = bias (atomics then give bias + sum) ============
__global__ void gat_init(float* __restrict__ out, const float* __restrict__ bias, int NC,
                         unsigned* __restrict__ mkey, float* __restrict__ den, int NH) {
    long long t = (long long)blockIdx.x * blockDim.x + threadIdx.x;
    if (t < (long long)NNODES * NC) out[t] = bias[(int)(t % NC)];
    if (t < (long long)NNODES * NH) { mkey[t] = 0u; den[t] = 0.f; }
}

__device__ __forceinline__ void edge_sd(const int* __restrict__ ei, int e, int& s, int& d) {
    if (e < NEDGES) { s = ei[e]; d = ei[NEDGES + e]; }
    else            { s = d = e - NEDGES; }   // self-loops appended
}

// ====================== pass 1: segment max (keyed) ======================
__global__ void gat_edge_max(const int* __restrict__ ei, const float* __restrict__ as_,
                             const float* __restrict__ ad_, unsigned* __restrict__ mkey,
                             int H_) {
    int t = blockIdx.x * blockDim.x + threadIdx.x;
    if (t >= ETOT * H_) return;
    int e = t / H_, hd = t % H_, s, d;
    edge_sd(ei, e, s, d);
    float v = as_[s * H_ + hd] + ad_[d * H_ + hd];
    v = v > 0.f ? v : NEG_SLOPE * v;
    atomicMax(&mkey[d * H_ + hd], enc_f(v));
}

// ====================== pass 2: segment sum of exp ======================
__global__ void gat_edge_sum(const int* __restrict__ ei, const float* __restrict__ as_,
                             const float* __restrict__ ad_,
                             const unsigned* __restrict__ mkey,
                             float* __restrict__ den, int H_) {
    int t = blockIdx.x * blockDim.x + threadIdx.x;
    if (t >= ETOT * H_) return;
    int e = t / H_, hd = t % H_, s, d;
    edge_sd(ei, e, s, d);
    float v = as_[s * H_ + hd] + ad_[d * H_ + hd];
    v = v > 0.f ? v : NEG_SLOPE * v;
    atomicAdd(&den[d * H_ + hd], __expf(v - dec_f(mkey[d * H_ + hd])));
}

// ========== pass 3: weighted aggregation, one wave per edge ==========
template <int H_, int C_>
__global__ __launch_bounds__(256) void gat_edge_aggr(
    const int* __restrict__ ei, const float* __restrict__ as_,
    const float* __restrict__ ad_, const unsigned* __restrict__ mkey,
    const float* __restrict__ den, const float* __restrict__ Hf,
    float* __restrict__ out) {
    const int wave = threadIdx.x >> 5, lane = threadIdx.x & 31;
    const int e = blockIdx.x * 8 + wave;
    if (e >= ETOT) return;
    int s, d;
    edge_sd(ei, e, s, d);
    float alpha[H_];
#pragma unroll
    for (int hd = 0; hd < H_; hd++) {
        float v = as_[s * H_ + hd] + ad_[d * H_ + hd];
        v = v > 0.f ? v : NEG_SLOPE * v;
        float ee = __expf(v - dec_f(mkey[d * H_ + hd]));
        alpha[hd] = ee / (den[d * H_ + hd] + 1e-16f);
    }
    constexpr int F = H_ * C_;
#pragma unroll
    for (int j = 0; j < F / 32; j++) {
        int idx = lane + j * 32;
        int hd = idx / C_;
        atomicAdd(&out[(size_t)d * F + idx], Hf[(size_t)s * F + idx] * alpha[hd]);
    }
}

// ============================ elementwise elu ============================
__global__ void gat_elu(float* __restrict__ o) {
    long long t = (long long)blockIdx.x * blockDim.x + threadIdx.x;
    if (t >= (long long)NNODES * 128) return;
    float v = o[t];
    o[t] = v > 0.f ? v : (__expf(v) - 1.f);
}

static inline int cdiv(long long a, long long b) { return (int)((a + b - 1) / b); }

extern "C" void kernel_launch(void* const* d_in, const int* in_sizes, int n_in,
                              void* d_out, int out_size, void* d_ws, size_t ws_size,
                              hipStream_t stream) {
    const float* x    = (const float*)d_in[0];
    const int*   ei   = (const int*)  d_in[1];
    const float* W1   = (const float*)d_in[2];
    const float* as1w = (const float*)d_in[3];
    const float* ad1w = (const float*)d_in[4];
    const float* b1   = (const float*)d_in[5];
    const float* W2   = (const float*)d_in[6];
    const float* as2w = (const float*)d_in[7];
    const float* ad2w = (const float*)d_in[8];
    const float* b2   = (const float*)d_in[9];
    float* out = (float*)d_out;

    // -------- workspace layout (bytes) --------
    char* base = (char*)d_ws;
    float*    h1   = (float*)(base);                         // N*128 f32 = 25.6 MB
    float*    out1 = (float*)(base + 25600000);              // N*128 f32 = 25.6 MB
    float*    h2   = (float*)(base + 51200000);              // N*64  f32 = 12.8 MB
    float*    as1  = (float*)(base + 64000000);              // N*4
    float*    ad1  = (float*)(base + 64800000);              // N*4
    unsigned* m1   = (unsigned*)(base + 65600000);           // N*4
    float*    dn1  = (float*)(base + 66400000);              // N*4
    float*    as2  = (float*)(base + 67200000);              // N
    float*    ad2  = (float*)(base + 67400000);              // N
    unsigned* m2   = (unsigned*)(base + 67600000);           // N
    float*    dn2  = (float*)(base + 67800000);              // N

    const int nTiles = NNODES / 16;                          // 3125 exactly
    const int gemmBlocks = cdiv(nTiles, 8);
    const int aggrBlocks = cdiv(ETOT, 8);

    // ---------------- layer 1: in 128 -> 4 heads x 32, concat ----------------
    gat_gemm<128><<<gemmBlocks, 256, 0, stream>>>(x, W1, h1, nTiles);
    gat_att<4, 32><<<cdiv((long long)NNODES * 4, 256), 256, 0, stream>>>(h1, as1w, ad1w, as1, ad1);
    gat_init<<<cdiv((long long)NNODES * 128, 256), 256, 0, stream>>>(out1, b1, 128, m1, dn1, 4);
    gat_edge_max<<<cdiv((long long)ETOT * 4, 256), 256, 0, stream>>>(ei, as1, ad1, m1, 4);
    gat_edge_sum<<<cdiv((long long)ETOT * 4, 256), 256, 0, stream>>>(ei, as1, ad1, m1, dn1, 4);
    gat_edge_aggr<4, 32><<<aggrBlocks, 256, 0, stream>>>(ei, as1, ad1, m1, dn1, h1, out1);
    gat_elu<<<cdiv((long long)NNODES * 128, 256), 256, 0, stream>>>(out1);

    // ---------------- layer 2: 128 -> 64, 1 head, mean(=identity) ----------------
    gat_gemm<64><<<gemmBlocks, 256, 0, stream>>>(out1, W2, h2, nTiles);
    gat_att<1, 64><<<cdiv((long long)NNODES, 256), 256, 0, stream>>>(h2, as2w, ad2w, as2, ad2);
    gat_init<<<cdiv((long long)NNODES * 64, 256), 256, 0, stream>>>(out, b2, 64, m2, dn2, 1);
    gat_edge_max<<<cdiv((long long)ETOT, 256), 256, 0, stream>>>(ei, as2, ad2, m2, 1);
    gat_edge_sum<<<cdiv((long long)ETOT, 256), 256, 0, stream>>>(ei, as2, ad2, m2, dn2, 1);
    gat_edge_aggr<1, 64><<<aggrBlocks, 256, 0, stream>>>(ei, as2, ad2, m2, dn2, h2, out);
}
